// Decoder_87857851007740
// MI455X (gfx1250) — matrix operand order repaired and checked by reading, not compiled, not run on hardware
//
#include <hip/hip_runtime.h>

// ---------------------------------------------------------------------------
// Problem constants (match the reference)
// ---------------------------------------------------------------------------
enum : int {
    Bb = 128, Tt = 256, Ll = 512, Hh = 512, Cc = 95, MAXLEN = 26,
    G3H = 3 * Hh,            // 1536
    LH  = Ll + Hh            // 1024
};

typedef __attribute__((ext_vector_type(16))) __bf16 bf16x16;
typedef __attribute__((ext_vector_type(8)))  float  floatx8;

// ---------------------------------------------------------------------------
// Weight pre-pack: convert f32 weight matrix into bf16 "fragment-major" layout
// so each lane's 16-element B fragment (32x16 tile, contiguous k order used here:
// K = (lane>>4)*16 + e, N = nt*16 + (lane&15)) is one contiguous 32-byte run.
//   BT=false: source stored [K,N] (ldb = N row stride)
//   BT=true : source stored [N,K] (ldb = K row stride)   (for A @ B^T)
// Packed index: (((kt * ntilesN) + nt) * 32 + lane) * 16 + e
// ---------------------------------------------------------------------------
template <bool BT>
__global__ __launch_bounds__(256) void pack_b_kernel(
    const float* __restrict__ B, __bf16* __restrict__ Bpk,
    int N, int K, int ldb)
{
    const int idx   = (int)(blockIdx.x * 256 + threadIdx.x);
    const int total = (K >> 5) * (N >> 4) * 512;
    if (idx >= total) return;
    const int e      = idx & 15;
    const int lane   = (idx >> 4) & 31;
    const int tile   = idx >> 9;
    const int ntiles = N >> 4;
    const int nt     = tile % ntiles;
    const int kt     = tile / ntiles;
    const int n = nt * 16 + (lane & 15);
    const int k = kt * 32 + (lane >> 4) * 16 + e;
    Bpk[idx] = (__bf16)(BT ? B[(size_t)n * ldb + k] : B[(size_t)k * ldb + n]);
}

// ---------------------------------------------------------------------------
// WMMA GEMM:  C[M,N] = A[M,K] (f32) * Bpacked (bf16 fragment-major) + bias[N]
// One wave owns a 16x64 strip of C (4 N-tiles; A fragment reused 4x/K-step).
// All 4 B fragments are loaded into distinct registers before the 4 WMMAs so
// the loads form one clause and the WMMAs issue back-to-back.
// A fragment (16-bit A 16x32 layout) loaded as 4x float4 per lane:
//   lane needs the contiguous K run [hi*16, hi*16+16) (same k order as packed B).
// Dims: M %16 == 0, N %64 == 0, K %32 == 0. Tile ids wave-uniform -> EXEC all-1.
// ---------------------------------------------------------------------------
__global__ __launch_bounds__(128) void gemm_bf16pk_wmma(
    const float* __restrict__ A, const __bf16* __restrict__ Bpk,
    const float* __restrict__ bias, float* __restrict__ Cm,
    int M, int N, int K, int lda, int ldc)
{
    const int wave  = (int)(threadIdx.x >> 5);
    const int lane  = (int)(threadIdx.x & 31);
    const int tn4   = N >> 6;                       // groups of 4 col-tiles
    const int wtile = (int)blockIdx.x * 4 + wave;
    const int tm    = (wtile / tn4) << 4;
    const int tn    = (wtile % tn4) << 6;
    if (tm >= M) return;

    const int hi     = lane >> 4;
    const int l16    = lane & 15;
    const int ntiles = N >> 4;

    floatx8 acc0 = {}, acc1 = {}, acc2 = {}, acc3 = {};

    // per-lane A base: row (tm + l16), starting at K offset hi*16
    const float* ar = A + (size_t)(tm + l16) * lda + hi * 16;
    // per-lane B base for this wave's 4 N-tiles at kt=0; advance by ntiles*512/kt
    const __bf16* bp = Bpk + ((size_t)(tn >> 4) * 32 + lane) * 16;
    const size_t bstep = (size_t)ntiles * 512;      // elements per K-tile row

    const int nkt = K >> 5;
    for (int kt = 0; kt < nkt; ++kt) {
        const int k0 = kt << 5;
        if (kt + 1 < nkt) __builtin_prefetch(ar + k0 + 32, 0, 3);

        // ---- A fragment: 4x b128 loads, compile-time placement ----
        const float4 p0 = *(const float4*)(ar + k0);
        const float4 p1 = *(const float4*)(ar + k0 + 4);
        const float4 p2 = *(const float4*)(ar + k0 + 8);
        const float4 p3 = *(const float4*)(ar + k0 + 12);

        // ---- all 4 B fragments up front (one clause, distinct registers) ----
        const __bf16* bk = bp + (size_t)kt * bstep;
        const bf16x16 b0 = *(const bf16x16*)(bk);
        const bf16x16 b1 = *(const bf16x16*)(bk + 512);
        const bf16x16 b2 = *(const bf16x16*)(bk + 1024);
        const bf16x16 b3 = *(const bf16x16*)(bk + 1536);

        bf16x16 a;
        a[0]  = (__bf16)p0.x; a[1]  = (__bf16)p0.y;
        a[2]  = (__bf16)p0.z; a[3]  = (__bf16)p0.w;
        a[4]  = (__bf16)p1.x; a[5]  = (__bf16)p1.y;
        a[6]  = (__bf16)p1.z; a[7]  = (__bf16)p1.w;
        a[8]  = (__bf16)p2.x; a[9]  = (__bf16)p2.y;
        a[10] = (__bf16)p2.z; a[11] = (__bf16)p2.w;
        a[12] = (__bf16)p3.x; a[13] = (__bf16)p3.y;
        a[14] = (__bf16)p3.z; a[15] = (__bf16)p3.w;

        acc0 = __builtin_amdgcn_wmma_f32_16x16x32_bf16(
            false, a, false, b0, (short)0, acc0, false, false);
        acc1 = __builtin_amdgcn_wmma_f32_16x16x32_bf16(
            false, a, false, b1, (short)0, acc1, false, false);
        acc2 = __builtin_amdgcn_wmma_f32_16x16x32_bf16(
            false, a, false, b2, (short)0, acc2, false, false);
        acc3 = __builtin_amdgcn_wmma_f32_16x16x32_bf16(
            false, a, false, b3, (short)0, acc3, false, false);
    }

    // ---- store: 32-bit C/D 16x16 layout ----
    const int mb = tm + (hi << 3);
#pragma unroll
    for (int nt = 0; nt < 4; ++nt) {
        const floatx8 acc = (nt == 0) ? acc0 : (nt == 1) ? acc1
                          : (nt == 2) ? acc2 : acc3;
        const int n = tn + nt * 16 + l16;
        const float bv = bias ? bias[n] : 0.0f;
#pragma unroll
        for (int v = 0; v < 8; ++v)
            Cm[(size_t)(mb + v) * ldc + n] = acc[v] + bv;
    }
}

// ---------------------------------------------------------------------------
// e[b,t] = wo . tanh(xEmb[b,t,:] + sEmb[b,:]) + bo   (one wave per (b,t),
// float4 loads: 128 vec4 per 512-row)
// ---------------------------------------------------------------------------
__global__ __launch_bounds__(256) void attn_scores_kernel(
    const float* __restrict__ xEmb, const float* __restrict__ sEmb,
    const float* __restrict__ wo, const float* __restrict__ bo,
    float* __restrict__ e)
{
    const int gwave = (int)((blockIdx.x * blockDim.x + threadIdx.x) >> 5);
    const int lane  = (int)(threadIdx.x & 31);
    if (gwave >= Bb * Tt) return;
    const int b = gwave >> 8;                       // T = 256
    const float4* xr = (const float4*)(xEmb + (size_t)gwave * Hh);
    const float4* sr = (const float4*)(sEmb + (size_t)b * Hh);
    const float4* wr = (const float4*)wo;
    float acc = 0.0f;
#pragma unroll
    for (int it = 0; it < 4; ++it) {
        const int i = it * 32 + lane;
        const float4 xv = xr[i], sv = sr[i], wv = wr[i];
        acc += tanhf(xv.x + sv.x) * wv.x + tanhf(xv.y + sv.y) * wv.y
             + tanhf(xv.z + sv.z) * wv.z + tanhf(xv.w + sv.w) * wv.w;
    }
    for (int off = 16; off > 0; off >>= 1)
        acc += __shfl_xor(acc, off, 32);
    if (lane == 0) e[gwave] = acc + bo[0];
}

// ---------------------------------------------------------------------------
// softmax over T per batch row (block per b, 256 threads)
// ---------------------------------------------------------------------------
__global__ __launch_bounds__(256) void softmax_T_kernel(
    const float* __restrict__ e, float* __restrict__ aw)
{
    __shared__ float red[256];
    const int b = blockIdx.x, t = (int)threadIdx.x;
    const float v = e[b * Tt + t];
    red[t] = v; __syncthreads();
    for (int s = 128; s > 0; s >>= 1) {
        if (t < s) red[t] = fmaxf(red[t], red[t + s]);
        __syncthreads();
    }
    const float m = red[0]; __syncthreads();
    const float ex = __expf(v - m);
    red[t] = ex; __syncthreads();
    for (int s = 128; s > 0; s >>= 1) {
        if (t < s) red[t] += red[t + s];
        __syncthreads();
    }
    aw[b * Tt + t] = ex / red[0];
}

// ---------------------------------------------------------------------------
// ctx[b,:] = sum_t aw[b,t] * x[b,t,:]   (block per b, 128 threads, float4)
// ---------------------------------------------------------------------------
__global__ __launch_bounds__(128) void ctx_kernel(
    const float* __restrict__ aw, const float* __restrict__ x,
    float* __restrict__ ctx)
{
    const int b  = blockIdx.x;
    const int l4 = (int)threadIdx.x;                // 128 float4 per row
    const float4* xb = (const float4*)(x + (size_t)b * Tt * Ll);
    const float*  ab = aw + (size_t)b * Tt;
    float4 acc = {0.0f, 0.0f, 0.0f, 0.0f};
    for (int t = 0; t < Tt; ++t) {
        const float  a  = ab[t];
        const float4 xv = xb[(size_t)t * 128 + l4];
        acc.x += a * xv.x; acc.y += a * xv.y;
        acc.z += a * xv.z; acc.w += a * xv.w;
    }
    ((float4*)ctx)[(size_t)b * 128 + l4] = acc;
}

// ---------------------------------------------------------------------------
// inp[b, 0:512] = emb[y_prev[b]],  inp[b, 512:1024] = ctx[b]
// ---------------------------------------------------------------------------
__global__ __launch_bounds__(256) void build_inp_kernel(
    const float* __restrict__ emb, const int* __restrict__ targets,
    const float* __restrict__ ctx, float* __restrict__ inp, int step)
{
    const int b = blockIdx.x;
    const int y = (step == 0) ? Cc : targets[b * MAXLEN + (step - 1)];
    const float* er = emb + (size_t)y * Hh;
    const float* cr = ctx + (size_t)b * Ll;
    float* ir = inp + (size_t)b * LH;
    for (int j = (int)threadIdx.x; j < LH; j += 256)
        ir[j] = (j < Hh) ? er[j] : cr[j - Hh];
}

// ---------------------------------------------------------------------------
// GRU gate fuse: r,z,n -> h_new  (gate order r,z,n as in torch)
// ---------------------------------------------------------------------------
__global__ __launch_bounds__(256) void gru_combine_kernel(
    const float* __restrict__ gi, const float* __restrict__ gh,
    const float* __restrict__ h, float* __restrict__ hn)
{
    const int idx = (int)(blockIdx.x * blockDim.x + threadIdx.x);   // B*H
    const int b = idx >> 9, j = idx & (Hh - 1);
    const float* gib = gi + (size_t)b * G3H;
    const float* ghb = gh + (size_t)b * G3H;
    const float r = 1.0f / (1.0f + __expf(-(gib[j] + ghb[j])));
    const float z = 1.0f / (1.0f + __expf(-(gib[Hh + j] + ghb[Hh + j])));
    const float n = tanhf(gib[2 * Hh + j] + r * ghb[2 * Hh + j]);
    hn[idx] = (1.0f - z) * n + z * h[idx];
}

// ---------------------------------------------------------------------------
// out[step,b,:] = log_softmax(h[b,:] @ Wout + bout)   (block per b, 128 thr)
// ---------------------------------------------------------------------------
__global__ __launch_bounds__(128) void out_logsoftmax_kernel(
    const float* __restrict__ h, const float* __restrict__ Wout,
    const float* __restrict__ bout, float* __restrict__ out, int step)
{
    __shared__ float logits[96];
    __shared__ float red[128];
    const int b = blockIdx.x, c = (int)threadIdx.x;
    float v = -3.0e38f;
    if (c < Cc) {
        float acc = bout[c];
        const float* hb = h + (size_t)b * Hh;
#pragma unroll 4
        for (int k = 0; k < Hh; ++k)
            acc += hb[k] * Wout[(size_t)k * Cc + c];
        logits[c] = acc;
        v = acc;
    }
    red[c] = v; __syncthreads();
    for (int s = 64; s > 0; s >>= 1) {
        if (c < s) red[c] = fmaxf(red[c], red[c + s]);
        __syncthreads();
    }
    const float m = red[0]; __syncthreads();
    const float ex = (c < Cc) ? __expf(logits[c] - m) : 0.0f;
    red[c] = ex; __syncthreads();
    for (int s = 64; s > 0; s >>= 1) {
        if (c < s) red[c] += red[c + s];
        __syncthreads();
    }
    const float lse = m + logf(red[0]);
    if (c < Cc)
        out[((size_t)step * Bb + b) * Cc + c] = logits[c] - lse;
}

__global__ __launch_bounds__(256) void fill_zero_kernel(float* __restrict__ p, int n)
{
    const int i = (int)(blockIdx.x * blockDim.x + threadIdx.x);
    if (i < n) p[i] = 0.0f;
}

// ---------------------------------------------------------------------------
// Host-side orchestration (graph-capture safe: only launches on `stream`)
// ---------------------------------------------------------------------------
extern "C" void kernel_launch(void* const* d_in, const int* in_sizes, int n_in,
                              void* d_out, int out_size, void* d_ws, size_t ws_size,
                              hipStream_t stream)
{
    (void)in_sizes; (void)n_in; (void)out_size; (void)ws_size;

    const float* x    = (const float*)d_in[0];
    const int*   tgt  = (const int*)  d_in[1];
    const float* emb  = (const float*)d_in[2];
    const float* Wi   = (const float*)d_in[3];
    const float* bi   = (const float*)d_in[4];
    const float* Wsm  = (const float*)d_in[5];
    const float* bs   = (const float*)d_in[6];
    const float* wo   = (const float*)d_in[7];
    const float* bo   = (const float*)d_in[8];
    const float* W_ih = (const float*)d_in[9];
    const float* W_hh = (const float*)d_in[10];
    const float* b_ih = (const float*)d_in[11];
    const float* b_hh = (const float*)d_in[12];
    const float* Wout = (const float*)d_in[13];
    const float* bout = (const float*)d_in[14];
    float* out = (float*)d_out;

    // workspace layout (floats, then 32B-aligned bf16 packed weights)
    float* ws   = (float*)d_ws;
    float* xEmb = ws;                                  // B*T*H = 16,777,216
    float* h    = xEmb + (size_t)Bb * Tt * Hh;         // B*H
    float* hn   = h    + (size_t)Bb * Hh;              // B*H
    float* sEmb = hn   + (size_t)Bb * Hh;              // B*H
    float* e    = sEmb + (size_t)Bb * Hh;              // B*T
    float* aw   = e    + (size_t)Bb * Tt;              // B*T
    float* ctx  = aw   + (size_t)Bb * Tt;              // B*L
    float* inp  = ctx  + (size_t)Bb * Ll;              // B*(L+H)
    float* gi   = inp  + (size_t)Bb * LH;              // B*3H
    float* gh   = gi   + (size_t)Bb * G3H;             // B*3H
    __bf16* WiPk  = (__bf16*)(gh + (size_t)Bb * G3H);  // 512*512
    __bf16* WsPk  = WiPk  + (size_t)Ll * Hh;           // 512*512
    __bf16* WihPk = WsPk  + (size_t)Hh * Hh;           // 1536*1024
    __bf16* WhhPk = WihPk + (size_t)G3H * LH;          // 1536*512

    // h0 = 0
    fill_zero_kernel<<<(Bb * Hh + 255) / 256, 256, 0, stream>>>(h, Bb * Hh);

    // ---- pre-pack all weights into bf16 fragment-major (once per call) ----
    pack_b_kernel<false><<<(Ll * Hh) / 256, 256, 0, stream>>>(Wi,   WiPk,  Hh,  Ll, Hh);
    pack_b_kernel<false><<<(Hh * Hh) / 256, 256, 0, stream>>>(Wsm,  WsPk,  Hh,  Hh, Hh);
    pack_b_kernel<true ><<<(G3H * LH) / 256, 256, 0, stream>>>(W_ih, WihPk, G3H, LH, LH);
    pack_b_kernel<true ><<<(G3H * Hh) / 256, 256, 0, stream>>>(W_hh, WhhPk, G3H, Hh, Hh);

    // xEmb = x @ Wi + bi   [32768,512] x [512,512]
    {
        const int M = Bb * Tt, N = Hh;
        const int waves = (M / 16) * (N / 64);
        gemm_bf16pk_wmma<<<waves / 4, 128, 0, stream>>>(
            x, WiPk, bi, xEmb, M, N, Ll, Ll, Hh);
    }

    float* hcur = h;
    float* hnew = hn;
    for (int step = 0; step < MAXLEN; ++step) {
        // sEmb = h @ Ws + bs   [128,512] x [512,512]
        gemm_bf16pk_wmma<<<(8 * 8) / 4, 128, 0, stream>>>(
            hcur, WsPk, bs, sEmb, Bb, Hh, Hh, Hh, Hh);

        // e = wo . tanh(xEmb + sEmb) + bo
        attn_scores_kernel<<<(Bb * Tt * 32) / 256, 256, 0, stream>>>(
            xEmb, sEmb, wo, bo, e);

        // aw = softmax_T(e)
        softmax_T_kernel<<<Bb, Tt, 0, stream>>>(e, aw);

        // ctx = aw . x
        ctx_kernel<<<Bb, 128, 0, stream>>>(aw, x, ctx);

        // inp = [emb[y_prev], ctx]
        build_inp_kernel<<<Bb, 256, 0, stream>>>(emb, tgt, ctx, inp, step);

        // gi = inp @ W_ih^T + b_ih   [128,1024] x [1536,1024]^T
        gemm_bf16pk_wmma<<<(8 * 24) / 4, 128, 0, stream>>>(
            inp, WihPk, b_ih, gi, Bb, G3H, LH, LH, G3H);

        // gh = h @ W_hh^T + b_hh     [128,512] x [1536,512]^T
        gemm_bf16pk_wmma<<<(8 * 24) / 4, 128, 0, stream>>>(
            hcur, WhhPk, b_hh, gh, Bb, G3H, Hh, Hh, G3H);

        // h_new = GRU(gi, gh, h)
        gru_combine_kernel<<<(Bb * Hh) / 256, 256, 0, stream>>>(gi, gh, hcur, hnew);

        // out[step] = log_softmax(h_new @ Wout + bout)
        out_logsoftmax_kernel<<<Bb, 128, 0, stream>>>(hnew, Wout, bout, out, step);

        float* tmp = hcur; hcur = hnew; hnew = tmp;
    }
}
